// EdgeModel_14585708937338
// MI455X (gfx1250) — compile-verified, matrix operand order checked by reading, not run on hardware
//
#include <hip/hip_runtime.h>
#include <hip/hip_bf16.h>

typedef __attribute__((ext_vector_type(16))) __bf16 bf16x16;
typedef __attribute__((ext_vector_type(8)))  float  f32x8;
typedef __attribute__((ext_vector_type(4)))  float  f32x4;

#define D   128          // node/edge feature dim
#define K1  384          // 3*D
#define H   256          // hidden
#define O   128          // output
#define MTILE 64         // edges per block (4 M-tiles of 16)

static __device__ __forceinline__ unsigned short f2bf(float f) {
    unsigned u = __builtin_bit_cast(unsigned, f);
    u += 0x7fffu + ((u >> 16) & 1u);          // round-to-nearest-even
    return (unsigned short)(u >> 16);
}

// ---- Prologue: f32 [K][N] weights -> bf16 [N][K] (transposed) in workspace ----
__global__ void convert_w1_kernel(const float* __restrict__ w1, unsigned short* __restrict__ w1t) {
    int o = blockIdx.x * blockDim.x + threadIdx.x;        // over H*K1 = 98304
    if (o >= H * K1) return;
    int n = o / K1, k = o % K1;
    w1t[o] = f2bf(w1[k * H + n]);
}
__global__ void convert_w2_kernel(const float* __restrict__ w2, unsigned short* __restrict__ w2t) {
    int o = blockIdx.x * blockDim.x + threadIdx.x;        // over O*H = 32768
    if (o >= O * H) return;
    int n = o / H, k = o % H;
    w2t[o] = f2bf(w2[k * O + n]);
}

// ---- Fused edge-MLP: gather -> (feat@W1+b1).relu -> (@W2+b2) ----
__global__ void __launch_bounds__(256, 2)
edge_mlp_kernel(const float* __restrict__ x,
                const float* __restrict__ edge_attr,
                const unsigned short* __restrict__ w1t,   // bf16 [H][K1]
                const float* __restrict__ b1,
                const unsigned short* __restrict__ w2t,   // bf16 [O][H]
                const float* __restrict__ b2,
                const long long* __restrict__ ei,         // int64 [E][2]
                float* __restrict__ out,                  // f32 [E][O]
                int E) {
    __shared__ __align__(32) unsigned short featLds[MTILE * K1]; // 48 KB bf16
    __shared__ __align__(32) unsigned short hLds[MTILE * H];     // 32 KB bf16

    const int tid = threadIdx.x;
    const int e0  = blockIdx.x * MTILE;

    // ---- Gather + f32->bf16 convert into LDS feat tile (64 rows x 384) ----
    // 3 segments x (64 rows x 32 float4) = 6144 slots; 256 thr -> 8/seg
    #pragma unroll
    for (int seg = 0; seg < 3; ++seg) {
        #pragma unroll
        for (int i = 0; i < 8; ++i) {
            int linear = tid + i * 256;                   // 0..2047
            int r  = linear >> 5;                         // row 0..63
            int c4 = linear & 31;                         // float4 column
            int e  = e0 + r;
            f32x4 v = {0.f, 0.f, 0.f, 0.f};
            if (e < E) {
                if (seg == 2) {
                    // streamed once: non-temporal, keep L2 for x + weights
                    v = __builtin_nontemporal_load(
                        (const f32x4*)(edge_attr + (size_t)e * D + c4 * 4));
                } else {
                    const long long node = ei[2 * e + seg];
                    v = *(const f32x4*)(x + (size_t)node * D + c4 * 4);
                }
            }
            unsigned* dst = (unsigned*)&featLds[r * K1 + seg * D + c4 * 4];
            dst[0] = (unsigned)f2bf(v.x) | ((unsigned)f2bf(v.y) << 16);
            dst[1] = (unsigned)f2bf(v.z) | ((unsigned)f2bf(v.w) << 16);
        }
    }
    __syncthreads();

    const int lane = tid & 31;
    const int wv   = tid >> 5;          // wave 0..7
    const int lm   = lane & 15;         // 0..15
    const int koff = (lane >> 4) * 16;  // K sub-offset, contiguous k = 16h + s (same for A and B)

    // ---- Layer 1: feat(64x384) @ W1T -> h(64x256) ----
    // wave = 2 M-tiles (32 rows) x 4 N-tiles (64 cols): each B frag feeds 2 WMMAs
    {
        const int mp = wv >> 2;         // M-pair 0..1 (rows mp*32 .. mp*32+31)
        const int nq = wv & 3;          // 64-col quarter 0..3
        f32x8 acc[2][4] = {};
        #pragma unroll
        for (int kk = 0; kk < K1 / 32; ++kk) {
            const bf16x16 a0 =
                *(const bf16x16*)&featLds[(mp * 32 + lm) * K1 + kk * 32 + koff];
            const bf16x16 a1 =
                *(const bf16x16*)&featLds[(mp * 32 + 16 + lm) * K1 + kk * 32 + koff];
            #pragma unroll
            for (int nt = 0; nt < 4; ++nt) {
                const bf16x16 b = *(const bf16x16*)(w1t +
                    (size_t)(nq * 64 + nt * 16 + lm) * K1 + kk * 32 + koff);
                acc[0][nt] = __builtin_amdgcn_wmma_f32_16x16x32_bf16(
                    false, a0, false, b, (short)0, acc[0][nt], false, false);
                acc[1][nt] = __builtin_amdgcn_wmma_f32_16x16x32_bf16(
                    false, a1, false, b, (short)0, acc[1][nt], false, false);
            }
        }
        // bias + relu -> bf16 h tile in LDS
        #pragma unroll
        for (int mi = 0; mi < 2; ++mi) {
            const int mrow = mp * 32 + mi * 16 + (lane >> 4) * 8;
            #pragma unroll
            for (int nt = 0; nt < 4; ++nt) {
                const int col  = nq * 64 + nt * 16 + lm;
                const float bb = b1[col];
                #pragma unroll
                for (int v = 0; v < 8; ++v) {
                    float hv = acc[mi][nt][v] + bb;
                    hv = hv > 0.f ? hv : 0.f;
                    hLds[(mrow + v) * H + col] = f2bf(hv);
                }
            }
        }
    }
    __syncthreads();

    // ---- Layer 2: h(64x256) @ W2T -> out(64x128) ----
    // wave = 2 M-tiles x 2 N-tiles
    {
        const int mp = wv & 1;          // M-pair 0..1 (rows mp*32 .. mp*32+31)
        const int nd = wv >> 1;         // 32-col slab 0..3
        f32x8 acc[2][2] = {};
        #pragma unroll
        for (int kk = 0; kk < H / 32; ++kk) {
            const bf16x16 a0 =
                *(const bf16x16*)&hLds[(mp * 32 + lm) * H + kk * 32 + koff];
            const bf16x16 a1 =
                *(const bf16x16*)&hLds[(mp * 32 + 16 + lm) * H + kk * 32 + koff];
            #pragma unroll
            for (int nt = 0; nt < 2; ++nt) {
                const bf16x16 b = *(const bf16x16*)(w2t +
                    (size_t)(nd * 32 + nt * 16 + lm) * H + kk * 32 + koff);
                acc[0][nt] = __builtin_amdgcn_wmma_f32_16x16x32_bf16(
                    false, a0, false, b, (short)0, acc[0][nt], false, false);
                acc[1][nt] = __builtin_amdgcn_wmma_f32_16x16x32_bf16(
                    false, a1, false, b, (short)0, acc[1][nt], false, false);
            }
        }
        #pragma unroll
        for (int mi = 0; mi < 2; ++mi) {
            const int mrow = mp * 32 + mi * 16 + (lane >> 4) * 8;
            #pragma unroll
            for (int nt = 0; nt < 2; ++nt) {
                const int col  = nd * 32 + nt * 16 + lm;
                const float bb = b2[col];
                #pragma unroll
                for (int v = 0; v < 8; ++v) {
                    int e = e0 + mrow + v;
                    if (e < E) {
                        // output streamed once: non-temporal store
                        __builtin_nontemporal_store(acc[mi][nt][v] + bb,
                                                    &out[(size_t)e * O + col]);
                    }
                }
            }
        }
    }
}

extern "C" void kernel_launch(void* const* d_in, const int* in_sizes, int n_in,
                              void* d_out, int out_size, void* d_ws, size_t ws_size,
                              hipStream_t stream) {
    const float*     x    = (const float*)d_in[0];
    const float*     ea   = (const float*)d_in[1];
    const float*     W1   = (const float*)d_in[2];
    const float*     b1   = (const float*)d_in[3];
    const float*     W2   = (const float*)d_in[4];
    const float*     b2   = (const float*)d_in[5];
    const long long* ei   = (const long long*)d_in[6];
    float*           out  = (float*)d_out;

    const int E = in_sizes[6] / 2;

    unsigned short* w1t = (unsigned short*)d_ws;          // H*K1 bf16 = 192 KB
    unsigned short* w2t = w1t + H * K1;                   // O*H  bf16 =  64 KB

    convert_w1_kernel<<<(H * K1 + 255) / 256, 256, 0, stream>>>(W1, w1t);
    convert_w2_kernel<<<(O * H + 255) / 256, 256, 0, stream>>>(W2, w2t);

    int grid = (E + MTILE - 1) / MTILE;                   // 7813 blocks
    edge_mlp_kernel<<<grid, 256, 0, stream>>>(x, ea, w1t, b1, w2t, b2, ei, out, E);
}
